// HyperGANDownBlock_49795850830342
// MI455X (gfx1250) — compile-verified, matrix operand order checked by reading, not run on hardware
//
#include <hip/hip_runtime.h>

typedef __attribute__((ext_vector_type(16))) __bf16 v16bf;
typedef __attribute__((ext_vector_type(2)))  __bf16 v2bf;
typedef __attribute__((ext_vector_type(8)))  float  v8f;
typedef __attribute__((ext_vector_type(16))) float  v16f;
typedef __attribute__((ext_vector_type(4)))  float  f4;
typedef __attribute__((ext_vector_type(2)))  float  f2;

#define CIN   32
#define COUT  64
#define DI    128
#define HI    128
#define WI    32
#define DO_   64
#define HO    64
#define WO    32
#define NPIX  (DO_*HO*WO)        /* 131072 output pixels per sample  */
#define TAPS  27
#define NFRAG (TAPS*4)           /* 108 B-fragments (27 taps x 4 N-tiles) */
#define LDSW  (NFRAG*32*32)      /* 110592 B of pre-swizzled bf16 weights */
#define LDS_TOTAL (LDSW + 128*4) /* + 64 sum + 64 sumsq floats */

// ---------------------------------------------------------------- stats zero
__global__ void zero_stats_k(float* __restrict__ ws) {
    int i = threadIdx.x;
    if (i < 512) ws[i] = 0.f;
}

// ---------------------------------------------------------------- conv+WMMA
// grid (256, 4), block 256 (8 waves). blockIdx.y = sample.
// Each wave owns 4 consecutive 16x64 tiles (64 consecutive output pixels) and
// keeps all 16 f32 accumulators live; B fragments are read from LDS once per
// tap and reused across the 4 tiles. Tap offsets are compile-time immediates.
__global__ __launch_bounds__(256)
void conv_wmma_k(const float* __restrict__ x, const float* __restrict__ w,
                 float* __restrict__ y, float* __restrict__ stats) {
    extern __shared__ char smem[];
    float* lsum = (float*)(smem + LDSW);      // [0..63] sum, [64..127] sumsq

    const int tid  = threadIdx.x;
    const int b    = blockIdx.y;
    const int lane = tid & 31;
    const int wv   = tid >> 5;
    const int sel  = lane >> 4;               // half-wave select
    const int nloc = lane & 15;

    if (tid < 128) lsum[tid] = 0.f;

    // ---- stage per-sample weights into LDS as pre-swizzled bf16 B-fragments.
    // B 32x16 (KxN) fragment layout (mirrors documented A 16x32 layout):
    //   lanes 0-15 hold K {0..7,16..23}, lanes 16-31 hold K {8..15,24..31},
    //   column N = lane%16; 16 bf16 per lane stored contiguously (32 B).
    const float* wb = w + (size_t)b * (TAPS * CIN * COUT);
    for (int it = tid; it < TAPS * 16 * 16; it += 256) {
        int t   = it >> 8;                    // 256 (k-pair, n-quad) per tap
        int rem = it & 255;
        int k   = (rem >> 4) << 1;            // even k; pair (k, k+1)
        int n0  = (rem & 15) << 2;            // 4 consecutive couts
        f4 w0 = *(const f4*)(wb + t * 2048 + k * 64 + n0);
        f4 w1 = *(const f4*)(wb + t * 2048 + (k + 1) * 64 + n0);
        int f    = (t << 2) + (n0 >> 4);
        int kb   = k & 15;
        int sel2 = (kb >= 8) ? 1 : 0;
        int i0   = (k < 16) ? (k - (sel2 << 3)) : (k - 8 - (sel2 << 3));
        int lb   = (sel2 << 4) + (n0 & 15);
        char* base = smem + ((size_t)f << 10) + (size_t)i0 * 2;
#pragma unroll
        for (int u = 0; u < 4; ++u) {
            f2 pf = { w0[u], w1[u] };
            *(v2bf*)(base + (size_t)(lb + u) * 32) =
                __builtin_convertvector(pf, v2bf);  // packed bf16 cvt
        }
    }
    __syncthreads();

    const float* xb = x + (size_t)b * ((size_t)DI * HI * WI * CIN);
    float*       yb = y + (size_t)b * ((size_t)NPIX * COUT);

    const int tile0 = blockIdx.x * 32 + (wv << 2); // 4 consecutive tiles
    const int m0    = tile0 << 4;

    // per-tile coordinates, hoisted base pointers and edge predicates
    const float* basep[4];
    bool dE[4], hE[4], wLo[4], wHi[4];
#pragma unroll
    for (int q = 0; q < 4; ++q) {
        int m  = m0 + (q << 4) + nloc;
        int ow = m & (WO - 1);
        int oh = (m >> 5) & (HO - 1);
        int od = m >> 11;
        // input coords for tap (0,0,0) are (2*od, 2*oh, ow-1); fold the -1
        // into the constant tap offsets instead (kw term), keep base at iw=ow.
        basep[q] = xb + ((size_t)(((od * 2) * HI + oh * 2) * WI + ow) << 5)
                      + (sel << 3);
        dE[q]  = (od == DO_ - 1);  // id==128 only when od==63 && kd==2
        hE[q]  = (oh == HO  - 1);  // ih==128 only when oh==63 && kh==2
        wLo[q] = (ow == 0);        // iw==-1  only when ow==0  && kw==0
        wHi[q] = (ow == WI - 1);   // iw==32  only when ow==31 && kw==2
    }

    v8f acc[4][4];
#pragma unroll
    for (int q = 0; q < 4; ++q)
#pragma unroll
        for (int j = 0; j < 4; ++j) acc[q][j] = (v8f){};

#pragma unroll
    for (int t = 0; t < TAPS; ++t) {
        const int kd = t / 9, r9 = t - kd * 9;
        const int kh = r9 / 3, kw = r9 - kh * 3;
        // constant byte-offset (in floats) of this tap from the tile base
        const int toff = ((kd * HI + kh) * WI + (kw - 1)) * CIN;

        const char* fb = smem + ((size_t)t << 12) + (lane << 5);
        v16bf b0 = *(const v16bf*)(fb);
        v16bf b1 = *(const v16bf*)(fb + 1024);
        v16bf b2 = *(const v16bf*)(fb + 2048);
        v16bf b3 = *(const v16bf*)(fb + 3072);

#pragma unroll
        for (int q = 0; q < 4; ++q) {
            // guard folds to an AND of precomputed per-tile flags
            bool ok = !((kd == 2) && dE[q]) && !((kh == 2) && hE[q]) &&
                      !((kw == 0) && wLo[q]) && !((kw == 2) && wHi[q]);

            v16bf a = {};                     // zeros when out of bounds
            if (ok) {
                const float* p = basep[q] + toff;
                f4 q0 = *(const f4*)(p);
                f4 q1 = *(const f4*)(p + 4);
                f4 q2 = *(const f4*)(p + 16);
                f4 q3 = *(const f4*)(p + 20);
                v8f g0 = __builtin_shufflevector(q0, q1, 0,1,2,3,4,5,6,7);
                v8f g1 = __builtin_shufflevector(q2, q3, 0,1,2,3,4,5,6,7);
                v16f gg = __builtin_shufflevector(g0, g1,
                            0,1,2,3,4,5,6,7,8,9,10,11,12,13,14,15);
                a = __builtin_convertvector(gg, v16bf);  // packed bf16 cvts
            }
            acc[q][0] = __builtin_amdgcn_wmma_f32_16x16x32_bf16(false, a, false, b0, (short)0, acc[q][0], false, false);
            acc[q][1] = __builtin_amdgcn_wmma_f32_16x16x32_bf16(false, a, false, b1, (short)0, acc[q][1], false, false);
            acc[q][2] = __builtin_amdgcn_wmma_f32_16x16x32_bf16(false, a, false, b2, (short)0, acc[q][2], false, false);
            acc[q][3] = __builtin_amdgcn_wmma_f32_16x16x32_bf16(false, a, false, b3, (short)0, acc[q][3], false, false);
        }
    }

    // C/D layout: VGPR r, lanes 0-15: (M=r, N=lane); lanes 16-31: (M=8+r).
#pragma unroll
    for (int q = 0; q < 4; ++q) {
        float* yt = yb + ((size_t)(m0 + (q << 4) + (sel << 3)) * COUT) + nloc;
#pragma unroll
        for (int j = 0; j < 4; ++j) {
            float s = 0.f, s2 = 0.f;
#pragma unroll
            for (int r = 0; r < 8; ++r) {
                float v = acc[q][j][r];
                yt[(size_t)r * COUT + (j << 4)] = v;
                s  += v;
                s2 += v * v;
            }
            atomicAdd(&lsum[(j << 4) + nloc], s);
            atomicAdd(&lsum[64 + (j << 4) + nloc], s2);
        }
    }
    __syncthreads();
    if (tid < 64) {
        atomicAdd(&stats[(b << 6) + tid],       lsum[tid]);
        atomicAdd(&stats[256 + (b << 6) + tid], lsum[64 + tid]);
    }
}

// ---------------------------------------------------------------- finalize
__global__ void finalize_stats_k(const float* __restrict__ gamma,
                                 const float* __restrict__ beta,
                                 float* __restrict__ ws) {
    int i = threadIdx.x;                       // 256 = 4 samples x 64 channels
    int c = i & 63;
    float inv  = 1.f / (float)NPIX;
    float mean = ws[i] * inv;
    float var  = ws[256 + i] * inv - mean * mean;
    float sc   = rsqrtf(var + 1e-5f) * gamma[c];
    ws[512 + i] = sc;
    ws[768 + i] = beta[c] - mean * sc;
}

// ---------------------------------------------------------------- norm+lrelu
__global__ __launch_bounds__(256)
void norm_lrelu_k(float* __restrict__ y, const float* __restrict__ ws) {
    size_t i = (size_t)blockIdx.x * 256 + threadIdx.x;
    size_t e = i << 2;                         // 4 consecutive channels
    int c  = (int)(e & 63);
    int b  = (int)(e >> 23);                   // NPIX*COUT = 2^23 per sample
    int si = (b << 6) + c;
    f4 v = *(const f4*)(y + e);
    f4 r;
#pragma unroll
    for (int k = 0; k < 4; ++k) {
        float t = v[k] * ws[512 + si + k] + ws[768 + si + k];
        r[k] = (t >= 0.f) ? t : 0.2f * t;
    }
    *(f4*)(y + e) = r;
}

// ---------------------------------------------------------------- launch
extern "C" void kernel_launch(void* const* d_in, const int* in_sizes, int n_in,
                              void* d_out, int out_size, void* d_ws, size_t ws_size,
                              hipStream_t stream) {
    const float* x     = (const float*)d_in[0];
    const float* w     = (const float*)d_in[1];
    const float* gamma = (const float*)d_in[2];
    const float* beta  = (const float*)d_in[3];
    float* out = (float*)d_out;
    float* ws  = (float*)d_ws;

    zero_stats_k<<<1, 512, 0, stream>>>(ws);

    dim3 grid(256, 4);                         // 256 WGs/sample x 4 samples
    conv_wmma_k<<<grid, 256, LDS_TOTAL, stream>>>(x, w, out, ws);

    finalize_stats_k<<<1, 256, 0, stream>>>(gamma, beta, ws);

    // 4*131072*64 floats / 4 per thread / 256 per block = 32768 blocks
    norm_lrelu_k<<<32768, 256, 0, stream>>>(out, ws);
}
